// RNN_23021024706898
// MI455X (gfx1250) — compile-verified, matrix operand order checked by reading, not run on hardware
//
#include <hip/hip_runtime.h>

#define H      512
#define BSZ    1024
#define LSEQ   64
#define NDEPTH 4
#define MROWS  16     // batch rows per workgroup (one WMMA M-tile)
#define HH     (H * H)

typedef __bf16 bf16_t;
typedef __bf16 v16bf __attribute__((ext_vector_type(16)));
typedef float  v8f   __attribute__((ext_vector_type(8)));

// ---------------------------------------------------------------------------
// Prep kernel: fp32 weights -> bf16, transposed to [N][K] so that B-fragments
// load with the same row-major pattern as A-fragments.
// ws layout (bf16): [0..3] = Wc[j]^T  ; [4..6] = Win_rest[j-1]^T
// ---------------------------------------------------------------------------
__global__ void wcvt_kernel(const float* __restrict__ Wc,
                            const float* __restrict__ Win_rest,
                            bf16_t* __restrict__ ws) {
  int idx = blockIdx.x * blockDim.x + threadIdx.x;
  if (idx >= 7 * HH) return;
  int m   = idx / HH;
  int rem = idx - m * HH;
  int n   = rem / H;        // output column (row of the transposed matrix)
  int k   = rem - n * H;    // reduction index
  const float* src = (m < 4) ? (Wc + m * HH) : (Win_rest + (m - 4) * HH);
  ws[idx] = (bf16_t)src[k * H + n];
}

// ---------------------------------------------------------------------------
// Fragment loader for 16x32 bf16 tiles, matching the CDNA5 16-bit A layout:
//   lane = r + 16*half; VGPR0-3 hold K = k0 + half*8 + {0..7},
//   VGPR4-7 hold K = k0 + 16 + half*8 + {0..7}.
// Works for both LDS (A) and global (B, pre-transposed) row-major sources.
// ---------------------------------------------------------------------------
__device__ __forceinline__ v16bf load_frag(const bf16_t* base, int row, int k0,
                                           int half) {
  const bf16_t* p = base + row * H + k0 + half * 8;
  union { v16bf v; uint4 q[2]; } u;
  u.q[0] = *(const uint4*)(p);
  u.q[1] = *(const uint4*)(p + 16);
  return u.v;
}

__device__ __forceinline__ float elu1(float v) {
  return (v > 0.0f) ? v : (__expf(v) - 1.0f);
}

// ---------------------------------------------------------------------------
// Persistent RNN kernel: one WG = 16 batch rows, 8 wave32s, full L=64 scan.
// Per layer: acc[nt] += state_j(16xK) @ WcT[j]   (K = H)
//            acc[nt] += h_{j-1}(16xK) @ WinT[j]  (j > 0, K = H)
// kt is the outer loop so one LDS A-fragment feeds 4 independent WMMA chains.
// ---------------------------------------------------------------------------
__global__ __launch_bounds__(256) void
RNN_23021024706898_kernel(const int* __restrict__ x,
                          const float* __restrict__ Win0,
                          const float* __restrict__ bc,
                          const float* __restrict__ Wout,
                          const float* __restrict__ bout,
                          const bf16_t* __restrict__ ws,
                          float* __restrict__ out) {
  __shared__ bf16_t stateB[NDEPTH][MROWS][H];   // 64 KB bf16 recurrent state
  __shared__ int    symLDS[MROWS];              // previous symbol per row

  const int tid  = threadIdx.x;
  const int lane = tid & 31;
  const int wave = tid >> 5;
  const int b0   = blockIdx.x * MROWS;

  const int r    = lane & 15;       // A row / B column within tile
  const int half = lane >> 4;       // K sub-group selector
  const int mb   = half * 8;        // C/D: rows mb..mb+7 in this lane

  // zero-init state, sym = -1 (initial r is all-zeros, not one-hot)
  {
    unsigned int* p = (unsigned int*)&stateB[0][0][0];
    const int total = NDEPTH * MROWS * H / 2;
    for (int i = tid; i < total; i += 256) p[i] = 0u;
    if (tid < MROWS) symLDS[tid] = -1;
  }
  __syncthreads();

  float lp_acc = 0.0f;  // wave 0 only: per-(row,class) log-prob accumulator

  for (int t = 0; t < LSEQ; ++t) {
    for (int j = 0; j < NDEPTH; ++j) {
      const bf16_t* WcT  = ws + j * HH;
      const bf16_t* WinT = ws + (3 + j) * HH;   // valid only for j >= 1
      // B base for this wave's first N-tile; tiles i are n0 = (wave*4+i)*16
      const int nbase = wave * 64;              // 4 tiles x 16 columns

      v8f acc[4];
      #pragma unroll
      for (int i = 0; i < 4; ++i) acc[i] = (v8f){};

      {
        const bf16_t* As = &stateB[j][0][0];
        const bf16_t* Bb = WcT + nbase * H;
        #pragma unroll 2
        for (int kt = 0; kt < H / 32; ++kt) {
          v16bf af = load_frag(As, r, kt * 32, half);   // shared by 4 tiles
          #pragma unroll
          for (int i = 0; i < 4; ++i) {
            v16bf bf = load_frag(Bb + i * 16 * H, r, kt * 32, half);
            acc[i] = __builtin_amdgcn_wmma_f32_16x16x32_bf16(
                false, af, false, bf, (short)0, acc[i], false, false);
          }
        }
      }
      if (j > 0) {
        const bf16_t* As = &stateB[j - 1][0][0];
        const bf16_t* Bb = WinT + nbase * H;
        #pragma unroll 2
        for (int kt = 0; kt < H / 32; ++kt) {
          v16bf af = load_frag(As, r, kt * 32, half);
          #pragma unroll
          for (int i = 0; i < 4; ++i) {
            v16bf bf = load_frag(Bb + i * 16 * H, r, kt * 32, half);
            acc[i] = __builtin_amdgcn_wmma_f32_16x16x32_bf16(
                false, af, false, bf, (short)0, acc[i], false, false);
          }
        }
      }
      __syncthreads();   // all reads of stateB[j] / stateB[j-1] complete

      // epilogue: bias (+ Win0 row-select for layer 0), ELU, store bf16
      #pragma unroll
      for (int i = 0; i < 4; ++i) {
        const int n = nbase + i * 16 + r;
        const float bias = bc[j * H + n];
        #pragma unroll
        for (int g = 0; g < 8; ++g) {
          float v = acc[i][g] + bias;
          if (j == 0) {
            int s = symLDS[mb + g];
            if (s >= 0) v += Win0[s * H + n];
          }
          stateB[j][mb + g][n] = (bf16_t)elu1(v);
        }
      }
      __syncthreads();   // stateB[j] now holds the new h for layer j+1
    }

    // output head: wave 0, 32 lanes = 16 rows x 2 classes
    if (wave == 0) {
      const int b = lane >> 1;
      const int c = lane & 1;
      const unsigned int* hrow = (const unsigned int*)&stateB[NDEPTH - 1][b][0];
      float dot = 0.0f;
      #pragma unroll 8
      for (int kk = 0; kk < H / 2; ++kk) {
        unsigned int u = hrow[kk];
        float h0 = __uint_as_float(u << 16);
        float h1 = __uint_as_float(u & 0xffff0000u);
        dot += h0 * Wout[(2 * kk) * 2 + c] + h1 * Wout[(2 * kk + 1) * 2 + c];
      }
      float logit = dot + bout[c];
      float other = __shfl_xor(logit, 1, 32);
      float mx    = fmaxf(logit, other);
      float lse   = mx + __logf(__expf(logit - mx) + __expf(other - mx));
      float logp  = logit - lse;
      int   xt    = x[(b0 + b) * LSEQ + t];
      if (c == xt) {
        if (!(logp == logp)) logp = -35.0f;   // nan_to_num
        lp_acc += logp;
      }
      if (c == 0) symLDS[b] = xt;             // r for the next step
    }
    __syncthreads();
  }

  if (wave == 0) {
    const int b = lane >> 1;
    float total = lp_acc + __shfl_xor(lp_acc, 1, 32);
    if ((lane & 1) == 0) out[b0 + b] = 0.5f * total;  // LOG_PROB_FACTOR
  }
}

// ---------------------------------------------------------------------------
extern "C" void kernel_launch(void* const* d_in, const int* in_sizes, int n_in,
                              void* d_out, int out_size, void* d_ws,
                              size_t ws_size, hipStream_t stream) {
  (void)in_sizes; (void)n_in; (void)out_size; (void)ws_size;
  const int*   x        = (const int*)d_in[0];
  const float* Win0     = (const float*)d_in[1];
  const float* Win_rest = (const float*)d_in[2];
  const float* Wc       = (const float*)d_in[3];
  const float* bc       = (const float*)d_in[4];
  const float* Wout     = (const float*)d_in[5];
  const float* bout     = (const float*)d_in[6];
  float*  out = (float*)d_out;
  bf16_t* wbf = (bf16_t*)d_ws;   // 7 * H * H bf16 = 3.5 MB

  const int total = 7 * HH;
  wcvt_kernel<<<(total + 255) / 256, 256, 0, stream>>>(Wc, Win_rest, wbf);
  RNN_23021024706898_kernel<<<BSZ / MROWS, 256, 0, stream>>>(
      x, Win0, bc, Wout, bout, wbf, out);
}
